// _SinkhornBase_83958020702540
// MI455X (gfx1250) — compile-verified, matrix operand order checked by reading, not run on hardware
//
#include <hip/hip_runtime.h>
#include <math.h>

#define EPSV     0.1f
#define INV_EPS  10.0f
#define MAX_ITER_ 50
#define THRESH_  0.05f

static constexpr int Bn = 32;
static constexpr int Nn = 1024;
static constexpr int Mn = 1024;
static constexpr int BN = Bn * Nn;                 // 32768
static constexpr int BM = Bn * Mn;                 // 32768
static constexpr long long NMl  = (long long)Nn * Mn;
static constexpr long long BNMl = (long long)Bn * NMl;

static constexpr int ROW_BLOCKS = BN / 128;        // 256 (128 rows per block)
static constexpr int COL_BLOCKS = BM / 128;        // 256 (128 cols per block)
static constexpr int ERRSZ      = ROW_BLOCKS;      // 256

// workspace layout (floats)
static constexpr int WS_U  = 0;
static constexpr int WS_V  = BN;
static constexpr int WS_E0 = BN + BM;
static constexpr int WS_E1 = WS_E0 + ERRSZ;
static constexpr int WS_RC = WS_E1 + ERRSZ;
static constexpr int RC_PER_B = 256;
static constexpr int WS_TOTAL = WS_RC + Bn * RC_PER_B;

typedef __attribute__((ext_vector_type(16))) _Float16     v16h;
typedef __attribute__((ext_vector_type(8)))  float        v8f;
typedef __attribute__((ext_vector_type(4)))  unsigned int v4u_;
typedef __attribute__((ext_vector_type(8)))  int          v8i_;
typedef __attribute__((ext_vector_type(4)))  int          v4i_;

// ---------------------------------------------------------------- init
__global__ void __launch_bounds__(256) sk_init(float* __restrict__ ws) {
  int i = blockIdx.x * blockDim.x + threadIdx.x;
  const int stride = gridDim.x * blockDim.x;
  for (; i < WS_TOTAL; i += stride) ws[i] = 0.0f;
}

// ------------------------------------------------- convergence gate
// Deterministic fixed-tree reduce of previous iteration's 256 per-block
// |du| partials. Block-uniform result.
__device__ __forceinline__ bool sk_gate(const float* __restrict__ errPrev,
                                        float* __restrict__ red, int iter) {
  if (iter == 0) return false;
  const int t = threadIdx.x;
  red[t] = errPrev[t];
  __syncthreads();
  #pragma unroll
  for (int s = 128; s > 0; s >>= 1) {
    if (t < s) red[t] += red[t + s];
    __syncthreads();
  }
  const bool skip = red[0] < (THRESH_ * (float)BN);
  __syncthreads();
  return skip;
}

// ------------------------------------------------- TDM 2D tile load
// D# describing a 128-row x 32-col f32 tile (row stride = 1024 elements)
// DMA'd from global into LDS at lds_off. 2D: groups 2/3 zero.
// This toolchain's builtin takes 6 args (extra int32x8 group, zero-filled).
__device__ __forceinline__ void tdm_load_tile(const float* gsrc,
                                              unsigned lds_off) {
  const unsigned long long ga = (unsigned long long)(uintptr_t)gsrc;
  v4u_ g0;
  g0[0] = 1u;                                   // count=1 (valid user D#)
  g0[1] = lds_off;                              // LDS byte address
  g0[2] = (unsigned)(ga & 0xffffffffu);         // global_addr[31:0]
  g0[3] = (unsigned)((ga >> 32) & 0x01ffffffu)  // global_addr[56:32]
        | (2u << 30);                           // type = 2 ("image")
  v8i_ g1;
  g1[0] = (int)(2u << 16);                      // data_size=4B, no pad/multicast
  g1[1] = (int)(1024u << 16);                   // tensor_dim0 = 1024 (lo16)
  g1[2] = (int)(1024u << 16);                   // dim0 hi | tensor_dim1 = 1024 (lo16)
  g1[3] = (int)(32u << 16);                     // dim1 hi | tile_dim0 = 32
  g1[4] = 128;                                  // tile_dim1 = 128, tile_dim2 = 0
  g1[5] = 1024;                                 // tensor_dim0_stride[31:0] = 1024
  g1[6] = 0;                                    // stride hi | dim1_stride lo
  g1[7] = 0;
  const v4i_ gz  = {0, 0, 0, 0};                // groups 2/3: 2D tensor
  const v8i_ gz8 = {0, 0, 0, 0, 0, 0, 0, 0};    // extra group (zero-filled)
  __builtin_amdgcn_tensor_load_to_lds(g0, g1, gz, gz, gz8, 0);
}

// ---------------------------------------------------------------- row pass
// u_new[b,i] = EPS * (log_mu - T0 - log( sum_j exp((v_j - C_ij)/EPS - T0) ))
// TDM double-buffers 128x32 tiles into LDS; each wave owns 16 rows and does
// the 1024-wide row sums as WMMA  E(16x32,f16) x ones  with f32 accumulate.
__global__ void __launch_bounds__(256) sk_row(const float* __restrict__ C,
                                              float* __restrict__ ws,
                                              int iter, float log_mu) {
  __shared__ float s_tileA[128 * 32];
  __shared__ float s_tileB[128 * 32];
  __shared__ float s_v[Mn];
  __shared__ float s_red[256];
  __shared__ float s_w[8];

  float*       u       = ws + WS_U;
  const float* v       = ws + WS_V;
  float*       errCur  = ws + ((iter & 1)       ? WS_E1 : WS_E0);
  const float* errPrev = ws + (((iter + 1) & 1) ? WS_E1 : WS_E0);

  const int t   = threadIdx.x;
  const int bid = blockIdx.x;

  if (sk_gate(errPrev, s_red, iter)) {          // frozen: sticky-zero partial
    if (t == 0) errCur[bid] = 0.0f;
    return;
  }

  const int wave = t >> 5;
  const int lane = t & 31;
  const long long rowBase = (long long)bid * 128;
  const float* gbase = C + rowBase * (long long)Mn;

  const unsigned ldsA = (unsigned)((uintptr_t)s_tileA & 0xffffffffull);
  const unsigned ldsB = (unsigned)((uintptr_t)s_tileB & 0xffffffffull);

  // kick off DMA of chunk 0 while we stage v and compute the shift
  if (wave == 0) tdm_load_tile(gbase, ldsA);

  const int b = bid >> 3;                        // 8 row-blocks per batch
  float vm = -3.0e38f;
  #pragma unroll
  for (int q = 0; q < Mn / 256; ++q) {
    const float val = v[b * Mn + q * 256 + t];
    s_v[q * 256 + t] = val;
    vm = fmaxf(vm, val);
  }
  s_red[t] = vm;
  __syncthreads();
  #pragma unroll
  for (int s = 128; s > 0; s >>= 1) {
    if (t < s) s_red[t] = fmaxf(s_red[t], s_red[t + s]);
    __syncthreads();
  }
  const float T0 = s_red[0] * INV_EPS;           // fixed stabilizing shift
  __syncthreads();

  const int m   = lane & 15;                     // matrix row within wave tile
  const int hlf = lane >> 4;

  v16h ones;
  #pragma unroll
  for (int e = 0; e < 16; ++e) ones[e] = (_Float16)1.0f;
  v8f acc = {0.f, 0.f, 0.f, 0.f, 0.f, 0.f, 0.f, 0.f};

  for (int chunk = 0; chunk < 32; ++chunk) {
    const int k0 = chunk * 32;
    if (wave == 0) {
      if (chunk + 1 < 32) {                      // prefetch next tile via TDM
        tdm_load_tile(gbase + (chunk + 1) * 32, (chunk & 1) ? ldsA : ldsB);
        __builtin_amdgcn_s_wait_tensorcnt(1);    // current tile complete
      } else {
        __builtin_amdgcn_s_wait_tensorcnt(0);
      }
    }
    __syncthreads();                             // tile visible to all waves

    const float* tile = (chunk & 1) ? s_tileB : s_tileA;
    // lanes m / m+16 together own matrix row m; rotated column access keeps
    // LDS banks conflict-free without padding (ones operand => K-slot
    // permutation does not change the row sum).
    const float* src = tile + (wave * 16 + m) * 32 + hlf * 16;
    const float* vv  = s_v + k0 + hlf * 16;
    v16h a;
    #pragma unroll
    for (int e = 0; e < 16; ++e) {
      const int cc = (e + m) & 15;
      const float x = (vv[cc] - src[cc]) * INV_EPS - T0;   // <= 0 by shift
      a[e] = (_Float16)__expf(x);
    }
    acc = __builtin_amdgcn_wmma_f32_16x16x32_f16(false, a, false, ones,
                                                 (short)0, acc, false, false);
    __syncthreads();                             // done reading before re-DMA
  }

  // D layout: lane 0 holds rows 0..7 sums in acc[0..7]; lane 16 rows 8..15
  float du = 0.0f;
  if (m == 0) {
    #pragma unroll
    for (int d = 0; d < 8; ++d) {
      const long long rg = rowBase + wave * 16 + hlf * 8 + d;
      const float s    = acc[d];
      const float unew = EPSV * (log_mu - (T0 + __logf(s)));
      const float uold = u[rg];
      u[rg] = unew;
      du += fabsf(unew - uold);                  // fixed order
    }
  }
  du += __shfl_down(du, 16);                     // lane0 += lane16
  if (lane == 0) s_w[wave] = du;
  __syncthreads();
  if (t == 0) {
    float tot = 0.0f;
    #pragma unroll
    for (int w = 0; w < 8; ++w) tot += s_w[w];   // fixed order
    errCur[bid] = tot;
  }
}

// ---------------------------------------------------------------- col pass
// v_new[b,j] = EPS * (log_nu - T0 - log( sum_i exp((u_i - C_ij)/EPS - T0) ))
// Wave handles 16 columns; column sums via WMMA  ones x E(32x16,f16).
__global__ void __launch_bounds__(256) sk_col(const float* __restrict__ C,
                                              float* __restrict__ ws,
                                              int iter, float log_nu) {
  __shared__ float s_u[Nn];
  __shared__ float s_red[256];

  const float* u       = ws + WS_U;
  float*       v       = ws + WS_V;
  const float* errPrev = ws + (((iter + 1) & 1) ? WS_E1 : WS_E0);

  const int t = threadIdx.x;
  if (sk_gate(errPrev, s_red, iter)) return;     // frozen: no-op

  const int bid = blockIdx.x;
  const int b   = bid >> 3;
  const int j0  = (bid & 7) * 128;

  float um = -3.0e38f;
  #pragma unroll
  for (int q = 0; q < Nn / 256; ++q) {
    const float val = u[b * Nn + q * 256 + t];
    s_u[q * 256 + t] = val;
    um = fmaxf(um, val);
  }
  s_red[t] = um;
  __syncthreads();
  #pragma unroll
  for (int s = 128; s > 0; s >>= 1) {
    if (t < s) s_red[t] = fmaxf(s_red[t], s_red[t + s]);
    __syncthreads();
  }
  const float T0 = s_red[0] * INV_EPS;
  __syncthreads();

  const int wave = t >> 5;
  const int lane = t & 31;
  const int n    = lane & 15;                    // matrix column within wave
  const int hlf  = lane >> 4;
  const int j    = j0 + wave * 16 + n;

  v16h ones;
  #pragma unroll
  for (int e = 0; e < 16; ++e) ones[e] = (_Float16)1.0f;
  v8f acc = {0.f, 0.f, 0.f, 0.f, 0.f, 0.f, 0.f, 0.f};

  const float* Cb = C + (long long)b * NMl;

  for (int i0 = 0; i0 < Nn; i0 += 32) {
    if (i0 + 32 < Nn)                            // CDNA5 global_prefetch_b8
      __builtin_prefetch(Cb + (long long)(i0 + 32 + hlf * 16) * Mn + j, 0, 1);
    v16h bm;
    #pragma unroll
    for (int r = 0; r < 16; ++r) {
      const int   i = i0 + hlf * 16 + r;
      const float c = Cb[(long long)i * Mn + j]; // 64B/half-wave coalesced
      const float x = (s_u[i] - c) * INV_EPS - T0;
      bm[r] = (_Float16)__expf(x);
    }
    acc = __builtin_amdgcn_wmma_f32_16x16x32_f16(false, ones, false, bm,
                                                 (short)0, acc, false, false);
  }

  // column-n sum replicated: acc[0] valid in both lane n and lane n+16
  if (lane < 16) {
    v[b * Mn + j] = EPSV * (log_nu - (T0 + __logf(acc[0])));
  }
}

// ---------------------------------------------------------------- finalize
// pi = exp((u_i + v_j - C)/EPS); copy cost; per-block partial of sum(pi*C)
__global__ void __launch_bounds__(256) sk_final(const float* __restrict__ C,
                                                float* __restrict__ ws,
                                                float* __restrict__ out) {
  __shared__ float s_red[256];
  const float* u = ws + WS_U;
  const float* v = ws + WS_V;
  float* rcpart  = ws + WS_RC;

  const int t = threadIdx.x;
  const long long base = (long long)blockIdx.x * 4096;
  const int b = blockIdx.x >> 8;                 // 256 blocks per batch

  float* pi_out = out + Bn;
  float* c_out  = out + Bn + BNMl;

  float acc = 0.0f;
  #pragma unroll
  for (int e = 0; e < 16; ++e) {
    const long long idx = base + e * 256 + t;
    const long long rg  = idx >> 10;             // global row b*N+i
    const int       j   = (int)(idx & 1023);
    const float c = C[idx];
    const float p = __expf((u[rg] + v[b * Mn + j] - c) * INV_EPS);
    pi_out[idx] = p;
    c_out[idx]  = c;
    acc += p * c;                                // fixed order
  }
  s_red[t] = acc;
  __syncthreads();
  #pragma unroll
  for (int s = 128; s > 0; s >>= 1) {
    if (t < s) s_red[t] += s_red[t + s];
    __syncthreads();
  }
  if (t == 0) rcpart[blockIdx.x] = s_red[0];
}

__global__ void __launch_bounds__(256) sk_rc(const float* __restrict__ ws,
                                             float* __restrict__ out) {
  __shared__ float s_red[256];
  const float* rcpart = ws + WS_RC;
  const int t = threadIdx.x;
  s_red[t] = rcpart[blockIdx.x * 256 + t];
  __syncthreads();
  #pragma unroll
  for (int s = 128; s > 0; s >>= 1) {
    if (t < s) s_red[t] += s_red[t + s];
    __syncthreads();
  }
  if (t == 0) out[blockIdx.x] = s_red[0];        // reduced_cost[b]
}

// ---------------------------------------------------------------- launch
extern "C" void kernel_launch(void* const* d_in, const int* in_sizes, int n_in,
                              void* d_out, int out_size, void* d_ws, size_t ws_size,
                              hipStream_t stream) {
  (void)in_sizes; (void)n_in; (void)out_size; (void)ws_size;
  const float* C  = (const float*)d_in[0];
  float*       ws = (float*)d_ws;
  float*       out = (float*)d_out;

  const float log_mu = logf(1.0f / (float)Nn + 1e-8f);
  const float log_nu = logf(1.0f / (float)Mn + 1e-8f);

  sk_init<<<64, 256, 0, stream>>>(ws);
  for (int it = 0; it < MAX_ITER_; ++it) {
    sk_row<<<ROW_BLOCKS, 256, 0, stream>>>(C, ws, it, log_mu);
    sk_col<<<COL_BLOCKS, 256, 0, stream>>>(C, ws, it, log_nu);
  }
  sk_final<<<Bn * RC_PER_B, 256, 0, stream>>>(C, ws, out);
  sk_rc<<<Bn, 256, 0, stream>>>(ws, out);
}
